// GPTLanguageModel_51762945851509
// MI455X (gfx1250) — compile-verified, matrix operand order checked by reading, not run on hardware
//
#include <hip/hip_runtime.h>

// ---------------- problem constants ----------------
static constexpr int B = 4, S = 1024, V = 32000, D = 1024, H = 16, L = 8;
static constexpr int HD = D / H;     // 64
static constexpr int FF = 4 * D;     // 4096
static constexpr int M_TOK = B * S;  // 4096

typedef __attribute__((ext_vector_type(16))) __bf16 v16bf;
typedef __attribute__((ext_vector_type(8)))  __bf16 v8bf;
typedef __attribute__((ext_vector_type(8)))  float  v8f;

// Async global->LDS staging (CDNA5 GLOBAL_LOAD_ASYNC_TO_LDS_B128, ASYNCcnt-tracked).
// Emitted via inline asm: the clang builtin exists but expects LangAS-qualified
// pointer types that are not spellable from HIP source.
#ifndef USE_ASYNC_COPY
#define USE_ASYNC_COPY 1
#endif

__device__ __forceinline__ void async_copy_b128(const void* gsrc, void* lds_dst) {
  unsigned loff = (unsigned)(unsigned long long)
      (__attribute__((address_space(3))) char*)lds_dst;  // LDS byte offset
  asm volatile("global_load_async_to_lds_b128 %0, %1, off"
               :: "v"(loff), "v"(gsrc) : "memory");
}

__device__ __forceinline__ unsigned short f32_to_bf16(float f) {
  unsigned int u = __float_as_uint(f);
  u += 0x7fffu + ((u >> 16) & 1u);  // round-to-nearest-even
  return (unsigned short)(u >> 16);
}

// ---------------- f32 -> bf16 convert ----------------
__global__ __launch_bounds__(256) void cvt_bf16_kernel(const float* __restrict__ in,
                                                       unsigned short* __restrict__ out,
                                                       long n) {
  long i0 = ((long)blockIdx.x * 256 + threadIdx.x) * 8;
  if (i0 + 8 <= n) {
    float4 a = *(const float4*)&in[i0];
    float4 b = *(const float4*)&in[i0 + 4];
    union { unsigned short us[8]; uint4 u4; } r;
    r.us[0] = f32_to_bf16(a.x); r.us[1] = f32_to_bf16(a.y);
    r.us[2] = f32_to_bf16(a.z); r.us[3] = f32_to_bf16(a.w);
    r.us[4] = f32_to_bf16(b.x); r.us[5] = f32_to_bf16(b.y);
    r.us[6] = f32_to_bf16(b.z); r.us[7] = f32_to_bf16(b.w);
    *(uint4*)&out[i0] = r.u4;
  } else {
    for (long i = i0; i < n; ++i) out[i] = f32_to_bf16(in[i]);
  }
}

// ---------------- embedding gather ----------------
__global__ __launch_bounds__(256) void embed_kernel(const int* __restrict__ idx,
                                                    const float* __restrict__ tok,
                                                    const float* __restrict__ pos,
                                                    float* __restrict__ x) {
  int row = blockIdx.x;  // b*S + s
  int s = row & (S - 1);
  long t = idx[row];
#pragma unroll
  for (int k = 0; k < D / 256; ++k) {
    int d = threadIdx.x + k * 256;
    x[(long)row * D + d] = tok[t * D + d] + pos[(long)s * D + d];
  }
}

// ---------------- LayerNorm -> bf16 ----------------
__global__ __launch_bounds__(256) void layernorm_bf16_kernel(const float* __restrict__ x,
                                                             const float* __restrict__ g,
                                                             const float* __restrict__ b,
                                                             unsigned short* __restrict__ out) {
  int row = blockIdx.x;
  const float* xr = x + (long)row * D;
  float v[4];
  float s = 0.f, s2 = 0.f;
#pragma unroll
  for (int k = 0; k < 4; ++k) {
    float t = xr[threadIdx.x + k * 256];
    v[k] = t; s += t; s2 += t * t;
  }
#pragma unroll
  for (int m = 16; m > 0; m >>= 1) { s += __shfl_xor(s, m, 32); s2 += __shfl_xor(s2, m, 32); }
  __shared__ float ps[8], ps2[8];
  int wave = threadIdx.x >> 5, lane = threadIdx.x & 31;
  if (lane == 0) { ps[wave] = s; ps2[wave] = s2; }
  __syncthreads();
  float ts = 0.f, ts2 = 0.f;
#pragma unroll
  for (int i = 0; i < 8; ++i) { ts += ps[i]; ts2 += ps2[i]; }
  float mean = ts * (1.f / D);
  float var = ts2 * (1.f / D) - mean * mean;
  float rstd = rsqrtf(var + 1e-5f);
#pragma unroll
  for (int k = 0; k < 4; ++k) {
    int i = threadIdx.x + k * 256;
    out[(long)row * D + i] = f32_to_bf16((v[k] - mean) * rstd * g[i] + b[i]);
  }
}

// ---------------- flash-style causal attention (chunked, LDS K/V staging) ----------------
// qkv: [B*S, H*(3*HD)] f32, per-head chunk = [q64|k64|v64]; out: bf16 [B*S, D].
// Block = 8 waves = 8 consecutive query rows of the SAME (b,h): K/V chunks staged once per block.
__global__ __launch_bounds__(256) void attn_kernel(const float* __restrict__ qkv,
                                                   unsigned short* __restrict__ out) {
  constexpr int CK = 32;   // keys per chunk (one per lane)
  constexpr int LVS = 68;  // LDS row stride in floats (16B aligned, conflict-light)
  __shared__ float lK[CK * LVS];
  __shared__ float lV[CK * LVS];
  __shared__ float lQ[8][HD];

  const int tid = threadIdx.x;
  const int lane = tid & 31;
  const int wave = tid >> 5;
  const long gid0 = (long)blockIdx.x * 8;  // first row id = b*H*S + h*S + qi0
  const int qi0 = (int)(gid0 & (S - 1));   // multiple of 8
  const int h = (int)((gid0 >> 10) & (H - 1));
  const int b = (int)(gid0 >> 14);
  const int qi = qi0 + wave;
  const float* base = qkv + (long)b * S * (3 * D) + (long)h * (3 * HD);

  // load scaled q for this wave (scale = HD^-0.5 folded in)
  const float* qr = base + (long)qi * (3 * D);
  lQ[wave][lane] = qr[lane] * 0.125f;
  lQ[wave][lane + 32] = qr[lane + 32] * 0.125f;
  __syncthreads();

  float m = -3.4e38f, lsum = 0.f, o0 = 0.f, o1 = 0.f;
  const int nchunk = (qi0 + 7) / CK + 1;  // covers keys 0..qi0+7
  for (int c = 0; c < nchunk; ++c) {
    const int j0 = c * CK;
    // cooperative stage: 32 rows x 64 floats for K and V (512 float4 each; 2 per thread)
#pragma unroll
    for (int t = 0; t < 2; ++t) {
      int cc = tid + t * 256;
      int r = cc >> 4;          // 16 float4 per row
      int col = (cc & 15) * 4;  // 0..60
      const float* src = base + (long)(j0 + r) * (3 * D) + HD;  // k row; v at +HD
      *(float4*)&lK[r * LVS + col] = *(const float4*)&src[col];
      *(float4*)&lV[r * LVS + col] = *(const float4*)&src[HD + col];
    }
    __syncthreads();
    if (j0 <= qi) {
      // score: this lane handles key j0+lane
      float s = 0.f;
#pragma unroll
      for (int d = 0; d < HD; d += 4) {
        float4 kk = *(const float4*)&lK[lane * LVS + d];
        s += lQ[wave][d] * kk.x + lQ[wave][d + 1] * kk.y +
             lQ[wave][d + 2] * kk.z + lQ[wave][d + 3] * kk.w;
      }
      float p = (j0 + lane <= qi) ? s : -3.4e38f;  // causal mask
      float pm = p;
#pragma unroll
      for (int mk = 16; mk > 0; mk >>= 1) pm = fmaxf(pm, __shfl_xor(pm, mk, 32));
      float mn = fmaxf(m, pm);
      float cf = __expf(m - mn);
      float e = __expf(p - mn);  // masked lanes underflow to 0
      float es = e;
#pragma unroll
      for (int mk = 16; mk > 0; mk >>= 1) es += __shfl_xor(es, mk, 32);
      lsum = lsum * cf + es;
      o0 *= cf; o1 *= cf;
      for (int jj = 0; jj < CK; ++jj) {
        float eb = __shfl(e, jj, 32);
        o0 += eb * lV[jj * LVS + lane];
        o1 += eb * lV[jj * LVS + lane + 32];
      }
      m = mn;
    }
    __syncthreads();
  }
  float inv = 1.f / lsum;
  unsigned short* o = out + ((long)b * S + qi) * D + h * HD;
  o[lane] = f32_to_bf16(o0 * inv);
  o[lane + 32] = f32_to_bf16(o1 * inv);
}

// ---------------- bf16 WMMA GEMM: C = A[M,K] * W[N,K]^T (+epilogue) ----------------
enum { GF_BIAS = 1, GF_RES = 2, GF_SILU = 4, GF_BF16 = 8 };

template <int FLAGS>
__global__ __launch_bounds__(256) void gemm_bf16_wmma(
    const unsigned short* __restrict__ A,   // [M,K] bf16
    const unsigned short* __restrict__ Bw,  // [N,K] bf16 (row = out feature)
    const float* __restrict__ bias,         // [N] or null
    const float* __restrict__ res,          // [M,N] or null (residual)
    void* __restrict__ Cout,                // [M,N] f32 or bf16
    int M, int N, int K) {
  constexpr int BM = 128, BN = 128, BK = 32, LDT = 40;  // padded LDS stride (bf16 elems)
  __shared__ unsigned short lA[2][BM * LDT];  // double-buffered
  __shared__ unsigned short lB[2][BN * LDT];

  const int tid = threadIdx.x;
  const int lane = tid & 31;
  const int wave = tid >> 5;
  const int wm0 = (wave >> 2) * 64;  // 2 waves along M, 4x 16-row tiles each
  const int wn0 = (wave & 3) * 32;   // 4 waves along N, 2x 16-col tiles each
  const long bm0 = (long)blockIdx.y * BM;
  const long bn0 = (long)blockIdx.x * BN;

  const int l16 = lane & 15;
  const int kb = (lane >> 4) * 8;  // per-half-wave K sub-chunk (ISA 16-bit A layout)

  // stage one 128x32 bf16 tile pair into buffer `buf` (512 x 16B chunks each)
  auto stage = [&](int buf, int k0) {
#pragma unroll
    for (int t = 0; t < 2; ++t) {
      int c = tid + t * 256;
      int r = c >> 2;
      int col = (c & 3) * 8;
#if USE_ASYNC_COPY
      async_copy_b128(&A[(bm0 + r) * K + k0 + col], &lA[buf][r * LDT + col]);
      async_copy_b128(&Bw[(bn0 + r) * K + k0 + col], &lB[buf][r * LDT + col]);
#else
      *(v8bf*)&lA[buf][r * LDT + col] = *(const v8bf*)&A[(bm0 + r) * K + k0 + col];
      *(v8bf*)&lB[buf][r * LDT + col] = *(const v8bf*)&Bw[(bn0 + r) * K + k0 + col];
#endif
    }
  };

  v8f acc[4][2];
  const v8f vzero = {0.f, 0.f, 0.f, 0.f, 0.f, 0.f, 0.f, 0.f};
#pragma unroll
  for (int mt = 0; mt < 4; ++mt)
#pragma unroll
    for (int nt = 0; nt < 2; ++nt) acc[mt][nt] = vzero;

  stage(0, 0);
#if USE_ASYNC_COPY
  asm volatile("s_wait_asynccnt 0x0" ::: "memory");
#endif
  __syncthreads();

  int buf = 0;
  for (int k0 = 0; k0 < K; k0 += BK) {
    if (k0 + BK < K) stage(buf ^ 1, k0 + BK);  // prefetch next tile into other buffer

    v16bf afrag[4], bfrag[2];
#pragma unroll
    for (int mt = 0; mt < 4; ++mt) {
      const int r = wm0 + mt * 16 + l16;
      v8bf lo = *(const v8bf*)&lA[buf][r * LDT + kb];
      v8bf hi = *(const v8bf*)&lA[buf][r * LDT + 16 + kb];
      afrag[mt] = __builtin_shufflevector(lo, hi, 0, 1, 2, 3, 4, 5, 6, 7, 8, 9, 10, 11, 12, 13, 14, 15);
    }
#pragma unroll
    for (int nt = 0; nt < 2; ++nt) {
      const int r = wn0 + nt * 16 + l16;
      v8bf lo = *(const v8bf*)&lB[buf][r * LDT + kb];
      v8bf hi = *(const v8bf*)&lB[buf][r * LDT + 16 + kb];
      bfrag[nt] = __builtin_shufflevector(lo, hi, 0, 1, 2, 3, 4, 5, 6, 7, 8, 9, 10, 11, 12, 13, 14, 15);
    }
#pragma unroll
    for (int mt = 0; mt < 4; ++mt)
#pragma unroll
      for (int nt = 0; nt < 2; ++nt)
        acc[mt][nt] = __builtin_amdgcn_wmma_f32_16x16x32_bf16(
            false, afrag[mt], false, bfrag[nt], (short)0, acc[mt][nt], false, false);

#if USE_ASYNC_COPY
    asm volatile("s_wait_asynccnt 0x0" ::: "memory");
#endif
    __syncthreads();
    buf ^= 1;
  }

  // epilogue (C layout: VGPR r -> M=r (lanes 0-15) / M=r+8 (lanes 16-31), N=lane&15)
  const int mh = (lane >> 4) * 8;
#pragma unroll
  for (int mt = 0; mt < 4; ++mt) {
#pragma unroll
    for (int nt = 0; nt < 2; ++nt) {
      const long colg = bn0 + wn0 + nt * 16 + l16;
#pragma unroll
      for (int r = 0; r < 8; ++r) {
        const long rowg = bm0 + wm0 + mt * 16 + mh + r;
        float vv = acc[mt][nt][r];
        if (FLAGS & GF_BIAS) vv += bias[colg];
        if (FLAGS & GF_RES) vv += res[rowg * N + colg];
        if (FLAGS & GF_SILU) vv = vv * (1.f / (1.f + __expf(-vv)));
        if (FLAGS & GF_BF16)
          ((unsigned short*)Cout)[rowg * N + colg] = f32_to_bf16(vv);
        else
          ((float*)Cout)[rowg * N + colg] = vv;
      }
    }
  }
}

// ---------------- host orchestration ----------------
extern "C" void kernel_launch(void* const* d_in, const int* in_sizes, int n_in,
                              void* d_out, int out_size, void* d_ws, size_t ws_size,
                              hipStream_t stream) {
  const int* idx = (const int*)d_in[0];
  const float* tok_emb = (const float*)d_in[1];
  const float* pos_emb = (const float*)d_in[2];
  const float* qkv_w = (const float*)d_in[3];
  const float* proj_w = (const float*)d_in[4];
  const float* proj_b = (const float*)d_in[5];
  const float* ln1_g = (const float*)d_in[6];
  const float* ln1_b = (const float*)d_in[7];
  const float* ln2_g = (const float*)d_in[8];
  const float* ln2_b = (const float*)d_in[9];
  const float* w1 = (const float*)d_in[10];
  const float* b1 = (const float*)d_in[11];
  const float* w2 = (const float*)d_in[12];
  const float* b2 = (const float*)d_in[13];
  const float* lnf_g = (const float*)d_in[14];
  const float* lnf_b = (const float*)d_in[15];
  const float* head_w = (const float*)d_in[16];
  const float* head_b = (const float*)d_in[17];

  // workspace layout (256B aligned offsets)
  char* ws = (char*)d_ws;
  unsigned short* wWb = (unsigned short*)(ws + 0);           // 65,536,000 B: bf16 weight scratch (max V*D)
  float* wX = (float*)(ws + 65536000);                       // 16,777,216 B: residual stream f32
  unsigned short* wHn = (unsigned short*)(ws + 82313216);    //  8,388,608 B: normalized act bf16
  float* wQKV = (float*)(ws + 90701824);                     // 50,331,648 B: qkv f32
  unsigned short* wAct = (unsigned short*)(ws + 141033472);  // 33,554,432 B: attn-out / FF1-out bf16

  dim3 blk(256);
  auto cvt = [&](const float* src, long n) {
    int grid = (int)((n + 8L * 256 - 1) / (8L * 256));
    cvt_bf16_kernel<<<grid, blk, 0, stream>>>(src, wWb, n);
  };

  embed_kernel<<<M_TOK, blk, 0, stream>>>(idx, tok_emb, pos_emb, wX);

  for (int l = 0; l < L; ++l) {
    // ---- attention block ----
    layernorm_bf16_kernel<<<M_TOK, blk, 0, stream>>>(wX, ln1_g + (long)l * D, ln1_b + (long)l * D, wHn);
    cvt(qkv_w + (long)l * 3 * D * D, (long)3 * D * D);
    gemm_bf16_wmma<0><<<dim3(3 * D / 128, M_TOK / 128), blk, 0, stream>>>(
        wHn, wWb, nullptr, nullptr, wQKV, M_TOK, 3 * D, D);
    attn_kernel<<<B * H * S / 8, blk, 0, stream>>>(wQKV, wAct);
    cvt(proj_w + (long)l * D * D, (long)D * D);
    gemm_bf16_wmma<GF_BIAS | GF_RES><<<dim3(D / 128, M_TOK / 128), blk, 0, stream>>>(
        wAct, wWb, proj_b + (long)l * D, wX, wX, M_TOK, D, D);
    // ---- MLP block ----
    layernorm_bf16_kernel<<<M_TOK, blk, 0, stream>>>(wX, ln2_g + (long)l * D, ln2_b + (long)l * D, wHn);
    cvt(w1 + (long)l * FF * D, (long)FF * D);
    gemm_bf16_wmma<GF_BIAS | GF_SILU | GF_BF16><<<dim3(FF / 128, M_TOK / 128), blk, 0, stream>>>(
        wHn, wWb, b1 + (long)l * FF, nullptr, wAct, M_TOK, FF, D);
    cvt(w2 + (long)l * D * FF, (long)D * FF);
    gemm_bf16_wmma<GF_BIAS | GF_RES><<<dim3(D / 128, M_TOK / 128), blk, 0, stream>>>(
        wAct, wWb, b2 + (long)l * D, wX, wX, M_TOK, D, FF);
  }

  // ---- final LN + LM head ----
  layernorm_bf16_kernel<<<M_TOK, blk, 0, stream>>>(wX, lnf_g, lnf_b, wHn);
  cvt(head_w, (long)V * D);
  gemm_bf16_wmma<GF_BIAS><<<dim3(V / 128, M_TOK / 128), blk, 0, stream>>>(
      wHn, wWb, head_b, nullptr, d_out, M_TOK, V, D);
}
